// MultiPointerGenerator_73023033966632
// MI455X (gfx1250) — compile-verified, matrix operand order checked by reading, not run on hardware
//
#include <hip/hip_runtime.h>

// ---------------------------------------------------------------------------
// MultiPointerGenerator for MI455X (gfx1250, wave32).
// B=16 T=24 D=512 V=32000, S = {32, 256, 64}.
// All GEMMs run on v_wmma_f32_16x16x32_bf16 (bf16 in, f32 accumulate).
// Double-buffered LDS staging; full tiles stream via
// global_load_async_to_lds_b128 (ASYNCcnt). Workspace requirement: ~55 MB.
// ---------------------------------------------------------------------------

#define Bn 16
#define Tn 24
#define Dn 512
#define Vn 32000
#define ROWS (Bn * Tn)              // 384
#define STOT (32 + 256 + 64)        // 352

typedef __attribute__((ext_vector_type(16))) __bf16 v16bf;
typedef __attribute__((ext_vector_type(8)))  float  v8f;
typedef int v4i __attribute__((vector_size(16)));

#ifndef __has_builtin
#define __has_builtin(x) 0
#endif
#if __has_builtin(__builtin_amdgcn_global_load_async_to_lds_b128) && \
    __has_builtin(__builtin_amdgcn_s_wait_asynccnt)
#define USE_ASYNC_LDS 1
#else
#define USE_ASYNC_LDS 0
#endif

__device__ __forceinline__ unsigned short f2bf(float f) {
    unsigned u = __float_as_uint(f);
    unsigned r = u + 0x7FFFu + ((u >> 16) & 1u);   // round-to-nearest-even
    return (unsigned short)(r >> 16);
}

#if USE_ASYNC_LDS
// global -> LDS async DMA, 16 bytes per lane, tracked by ASYNCcnt.
__device__ __forceinline__ void async_cp16(const unsigned short* g, unsigned short* l) {
    __builtin_amdgcn_global_load_async_to_lds_b128(
        (__attribute__((address_space(1))) v4i*)(uintptr_t)g,
        (__attribute__((address_space(3))) v4i*)(uintptr_t)l,
        0, 0);
}
#endif

// ------------------------- reductions (wave32) -----------------------------
__device__ __forceinline__ float warp_max(float v) {
    #pragma unroll
    for (int o = 16; o; o >>= 1) v = fmaxf(v, __shfl_xor(v, o, 32));
    return v;
}
__device__ __forceinline__ float warp_sum(float v) {
    #pragma unroll
    for (int o = 16; o; o >>= 1) v += __shfl_xor(v, o, 32);
    return v;
}
__device__ float block_max(float v) {
    __shared__ float s[32];
    int w = threadIdx.x >> 5, lane = threadIdx.x & 31;
    v = warp_max(v);
    if (lane == 0) s[w] = v;
    __syncthreads();
    int nw = blockDim.x >> 5;
    float r = (threadIdx.x < nw) ? s[threadIdx.x] : -INFINITY;
    if (w == 0) { r = warp_max(r); if (lane == 0) s[0] = r; }
    __syncthreads();
    r = s[0];
    __syncthreads();
    return r;
}
__device__ float block_sum(float v) {
    __shared__ float s[32];
    int w = threadIdx.x >> 5, lane = threadIdx.x & 31;
    v = warp_sum(v);
    if (lane == 0) s[w] = v;
    __syncthreads();
    int nw = blockDim.x >> 5;
    float r = (threadIdx.x < nw) ? s[threadIdx.x] : 0.0f;
    if (w == 0) { r = warp_sum(r); if (lane == 0) s[0] = r; }
    __syncthreads();
    r = s[0];
    __syncthreads();
    return r;
}

// ------------------------- fp32 -> bf16 convert ----------------------------
__global__ void cvt_bf16_kernel(const float* __restrict__ src,
                                unsigned short* __restrict__ dst, long n) {
    long i = (long)blockIdx.x * blockDim.x + threadIdx.x;
    long stride = (long)gridDim.x * blockDim.x;
    for (; i < n; i += stride) dst[i] = f2bf(src[i]);
}

// transpose + convert: src [R,C] f32 row-major -> dst [C,R] bf16 row-major
__global__ void tcvt_bf16_kernel(const float* __restrict__ src,
                                 unsigned short* __restrict__ dst, int R, int C) {
    __shared__ float tile[32][33];
    int c0 = blockIdx.x * 32, r0 = blockIdx.y * 32;
    for (int j = threadIdx.y; j < 32; j += 8) {
        int r = r0 + j, c = c0 + threadIdx.x;
        tile[j][threadIdx.x] = (r < R && c < C) ? src[(size_t)r * C + c] : 0.0f;
    }
    __syncthreads();
    for (int j = threadIdx.y; j < 32; j += 8) {
        int n = c0 + j, k = r0 + threadIdx.x;
        if (n < C && k < R) dst[(size_t)n * R + k] = f2bf(tile[threadIdx.x][j]);
    }
}

// ---------------------------------------------------------------------------
// Generic WMMA GEMM:  C[M,N] = scale * (A[M,K] * Bt[N,K]^T) + bias[N]
// A, Bt: bf16 row-major (K contiguous).  C: f32.  batch = blockIdx.z.
// 128 threads = 4 waves; block tile 64(M) x 64(N); K-step 32; K % 32 == 0.
// Wave w computes rows [w*16, w*16+16) x 64 columns (4 v8f accumulators).
// LDS rows are 48 halves (96 B) so every 8-half chunk is 16-byte aligned.
// ---------------------------------------------------------------------------
__device__ __forceinline__ void wmma_tile(const unsigned short (*__restrict__ As)[48],
                                          const unsigned short (*__restrict__ Bs)[48],
                                          int aRow, int kc, int lane15, v8f acc[4]) {
    v16bf a, b0, b1, b2, b3;
    ((uint4*)&a)[0]  = *(const uint4*)&As[aRow][kc];
    ((uint4*)&a)[1]  = *(const uint4*)&As[aRow][kc + 16];
    ((uint4*)&b0)[0] = *(const uint4*)&Bs[lane15][kc];
    ((uint4*)&b0)[1] = *(const uint4*)&Bs[lane15][kc + 16];
    ((uint4*)&b1)[0] = *(const uint4*)&Bs[16 + lane15][kc];
    ((uint4*)&b1)[1] = *(const uint4*)&Bs[16 + lane15][kc + 16];
    ((uint4*)&b2)[0] = *(const uint4*)&Bs[32 + lane15][kc];
    ((uint4*)&b2)[1] = *(const uint4*)&Bs[32 + lane15][kc + 16];
    ((uint4*)&b3)[0] = *(const uint4*)&Bs[48 + lane15][kc];
    ((uint4*)&b3)[1] = *(const uint4*)&Bs[48 + lane15][kc + 16];
    acc[0] = __builtin_amdgcn_wmma_f32_16x16x32_bf16(false, a, false, b0, (short)0, acc[0], false, false);
    acc[1] = __builtin_amdgcn_wmma_f32_16x16x32_bf16(false, a, false, b1, (short)0, acc[1], false, false);
    acc[2] = __builtin_amdgcn_wmma_f32_16x16x32_bf16(false, a, false, b2, (short)0, acc[2], false, false);
    acc[3] = __builtin_amdgcn_wmma_f32_16x16x32_bf16(false, a, false, b3, (short)0, acc[3], false, false);
}

__global__ void gemm_bt_wmma(const unsigned short* __restrict__ A,
                             const unsigned short* __restrict__ Bt,
                             float* __restrict__ C,
                             const float* __restrict__ bias,
                             int M, int N, int K,
                             int ldA, int ldB, int ldC,
                             long sA, long sB, long sC, float scale) {
    A  += (long)blockIdx.z * sA;
    Bt += (long)blockIdx.z * sB;
    C  += (long)blockIdx.z * sC;

    const int tid  = threadIdx.x;
    const int w    = tid >> 5;
    const int lane = tid & 31;
    const int lane15 = lane & 15;
    const int m0 = blockIdx.y * 64;
    const int n0 = blockIdx.x * 64;

    __shared__ unsigned short As[2][64][48];
    __shared__ unsigned short Bs[2][64][48];

    v8f acc[4];
    #pragma unroll
    for (int j = 0; j < 4; ++j)
        #pragma unroll
        for (int r = 0; r < 8; ++r) acc[j][r] = 0.0f;

    const int aRow = w * 16 + lane15;
    const int kc   = (lane < 16) ? 0 : 8;     // CDNA5 16-bit WMMA K-interleave

    // staging assignment: thread covers rows rA and rA+32, 8-half chunk c8
    const int rA = tid >> 2;
    const int c8 = (tid & 3) << 3;
    const int nk = K >> 5;                    // K % 32 == 0, nk >= 1

    if (m0 + 64 <= M && n0 + 64 <= N) {
        // ---------------- full-tile fast path (no guards) ----------------
        const unsigned short* pa0 = A  + (size_t)(m0 + rA) * ldA + c8;
        const unsigned short* pa1 = pa0 + (size_t)32 * ldA;
        const unsigned short* pb0 = Bt + (size_t)(n0 + rA) * ldB + c8;
        const unsigned short* pb1 = pb0 + (size_t)32 * ldB;
        int buf = 0;
#if USE_ASYNC_LDS
        // prime tile 0 via async global->LDS DMA
        async_cp16(pa0, &As[0][rA][c8]);
        async_cp16(pa1, &As[0][rA + 32][c8]);
        async_cp16(pb0, &Bs[0][rA][c8]);
        async_cp16(pb1, &Bs[0][rA + 32][c8]);
        pa0 += 32; pa1 += 32; pb0 += 32; pb1 += 32;
        // steady state: branch-free pipeline body (last iteration peeled)
        for (int t = 0; t < nk - 1; ++t) {
            async_cp16(pa0, &As[buf ^ 1][rA][c8]);
            async_cp16(pa1, &As[buf ^ 1][rA + 32][c8]);
            async_cp16(pb0, &Bs[buf ^ 1][rA][c8]);
            async_cp16(pb1, &Bs[buf ^ 1][rA + 32][c8]);
            pa0 += 32; pa1 += 32; pb0 += 32; pb1 += 32;
            // in-order completion: waiting to <=4 retires the previous tile
            __builtin_amdgcn_s_wait_asynccnt(4);
            __syncthreads();
            wmma_tile(As[buf], Bs[buf], aRow, kc, lane15, acc);
            __syncthreads();
            buf ^= 1;
        }
        __builtin_amdgcn_s_wait_asynccnt(0);
        __syncthreads();
        wmma_tile(As[buf], Bs[buf], aRow, kc, lane15, acc);
#else
        // register-double-buffered fallback (last iteration peeled)
        uint4 ra0 = *(const uint4*)pa0, ra1 = *(const uint4*)pa1;
        uint4 rb0 = *(const uint4*)pb0, rb1 = *(const uint4*)pb1;
        pa0 += 32; pa1 += 32; pb0 += 32; pb1 += 32;
        for (int t = 0; t < nk - 1; ++t) {
            *(uint4*)&As[buf][rA][c8]      = ra0;
            *(uint4*)&As[buf][rA + 32][c8] = ra1;
            *(uint4*)&Bs[buf][rA][c8]      = rb0;
            *(uint4*)&Bs[buf][rA + 32][c8] = rb1;
            __syncthreads();
            // next tile's loads overlap the WMMAs below
            ra0 = *(const uint4*)pa0; ra1 = *(const uint4*)pa1;
            rb0 = *(const uint4*)pb0; rb1 = *(const uint4*)pb1;
            pa0 += 32; pa1 += 32; pb0 += 32; pb1 += 32;
            wmma_tile(As[buf], Bs[buf], aRow, kc, lane15, acc);
            __syncthreads();
            buf ^= 1;
        }
        *(uint4*)&As[buf][rA][c8]      = ra0;
        *(uint4*)&As[buf][rA + 32][c8] = ra1;
        *(uint4*)&Bs[buf][rA][c8]      = rb0;
        *(uint4*)&Bs[buf][rA + 32][c8] = rb1;
        __syncthreads();
        wmma_tile(As[buf], Bs[buf], aRow, kc, lane15, acc);
#endif
    } else {
        // ---------------- guarded path (partial tiles; scores GEMM M=24) --
        const int gr0 = m0 + rA, gr1 = gr0 + 32;
        const int gn0 = n0 + rA, gn1 = gn0 + 32;
        const bool va0 = gr0 < M, va1 = gr1 < M;
        const bool vb0 = gn0 < N, vb1 = gn1 < N;
        const unsigned short* pa0 = A  + (size_t)gr0 * ldA + c8;
        const unsigned short* pa1 = A  + (size_t)gr1 * ldA + c8;
        const unsigned short* pb0 = Bt + (size_t)gn0 * ldB + c8;
        const unsigned short* pb1 = Bt + (size_t)gn1 * ldB + c8;
        const uint4 zz = make_uint4(0u, 0u, 0u, 0u);
        for (int t = 0; t < nk; ++t) {
            *(uint4*)&As[0][rA][c8]      = va0 ? *(const uint4*)pa0 : zz;
            *(uint4*)&As[0][rA + 32][c8] = va1 ? *(const uint4*)pa1 : zz;
            *(uint4*)&Bs[0][rA][c8]      = vb0 ? *(const uint4*)pb0 : zz;
            *(uint4*)&Bs[0][rA + 32][c8] = vb1 ? *(const uint4*)pb1 : zz;
            pa0 += 32; pa1 += 32; pb0 += 32; pb1 += 32;
            __syncthreads();
            wmma_tile(As[0], Bs[0], aRow, kc, lane15, acc);
            __syncthreads();
        }
    }

    // ---- epilogue: VGPR r holds M=r (lanes 0-15) / M=r+8 (lanes 16-31) ----
    const int rbase = m0 + w * 16 + ((lane < 16) ? 0 : 8);
    #pragma unroll
    for (int j = 0; j < 4; ++j) {
        int col = n0 + j * 16 + lane15;
        if (col < N) {
            float bb = bias ? bias[col] : 0.0f;
            #pragma unroll
            for (int r = 0; r < 8; ++r) {
                int row = rbase + r;
                if (row < M) C[(size_t)row * ldC + col] = acc[j][r] * scale + bb;
            }
        }
    }
}

// ------------------- masked softmax over S (in place) ----------------------
__global__ void attn_softmax_kernel(float* __restrict__ scores,       // [B*T, S]
                                    const int* __restrict__ text,     // [B, S]
                                    const unsigned char* __restrict__ mask, // [B,1,S]
                                    int S) {
    int row = blockIdx.x;
    int b = row / Tn;
    float* sc = scores + (size_t)row * S;
    const int* tx = text + (size_t)b * S;
    const unsigned char* mk = mask + (size_t)b * S;

    float lmax = -INFINITY;
    for (int s = threadIdx.x; s < S; s += blockDim.x) {
        bool m = (mk[s] != 0) && (tx[s] != 0);
        float v = m ? sc[s] : -1e9f;
        sc[s] = v;
        lmax = fmaxf(lmax, v);
    }
    float mx = block_max(lmax);
    float lsum = 0.0f;
    for (int s = threadIdx.x; s < S; s += blockDim.x) {
        float e = __expf(sc[s] - mx);
        sc[s] = e;
        lsum += e;
    }
    float inv = 1.0f / block_sum(lsum);
    for (int s = threadIdx.x; s < S; s += blockDim.x) sc[s] *= inv;
}

// -------------------- text_vec = attn[T,S] @ enc[S,D] ----------------------
__global__ void text_vec_kernel(const float* __restrict__ attn,  // [B*T, S]
                                const float* __restrict__ enc,   // [B, S, D]
                                float* __restrict__ vout,        // [B*T, D]
                                int S) {
    int row = blockIdx.x;
    int b = row / Tn;
    const float* ar = attn + (size_t)row * S;
    const float* eb = enc + (size_t)b * S * Dn;
    int d = threadIdx.x;                 // 256 threads -> 2 d each
    float a0 = 0.0f, a1 = 0.0f;
    for (int s = 0; s < S; ++s) {
        float a = ar[s];
        a0 += a * eb[(size_t)s * Dn + d];
        a1 += a * eb[(size_t)s * Dn + d + 256];
    }
    vout[(size_t)row * Dn + d]       = a0;
    vout[(size_t)row * Dn + d + 256] = a1;
}

// -------------------- mixture gate: softmax(gen_vec @ W + b) ---------------
__global__ void mix_kernel(const float* __restrict__ logits,
                           const float* __restrict__ enc_tgt,
                           const float* __restrict__ v0,
                           const float* __restrict__ v1,
                           const float* __restrict__ v2,
                           const float* __restrict__ W,   // [5D, 4]
                           const float* __restrict__ bg,  // [4]
                           float* __restrict__ mixout) {  // [B*T, 4]
    int row = blockIdx.x;
    int warp = threadIdx.x >> 5;   // 0..3 : output column
    int lane = threadIdx.x & 31;
    const float* secs[5] = { logits  + (size_t)row * Dn, enc_tgt + (size_t)row * Dn,
                             v0 + (size_t)row * Dn, v1 + (size_t)row * Dn,
                             v2 + (size_t)row * Dn };
    float acc = 0.0f;
    for (int sec = 0; sec < 5; ++sec) {
        const float* f = secs[sec];
        int base = sec * Dn;
        for (int e = lane; e < Dn; e += 32)
            acc += f[e] * W[(size_t)(base + e) * 4 + warp];
    }
    #pragma unroll
    for (int o = 16; o; o >>= 1) acc += __shfl_down(acc, o, 32);
    __shared__ float col[4];
    if (lane == 0) col[warp] = acc + bg[warp];
    __syncthreads();
    if (threadIdx.x == 0) {
        float m = fmaxf(fmaxf(col[0], col[1]), fmaxf(col[2], col[3]));
        float e0 = __expf(col[0] - m), e1 = __expf(col[1] - m);
        float e2 = __expf(col[2] - m), e3 = __expf(col[3] - m);
        float inv = 1.0f / (e0 + e1 + e2 + e3);
        mixout[(size_t)row * 4 + 0] = e0 * inv;
        mixout[(size_t)row * 4 + 1] = e1 * inv;
        mixout[(size_t)row * 4 + 2] = e2 * inv;
        mixout[(size_t)row * 4 + 3] = e3 * inv;
    }
}

// ---------------------------------------------------------------------------
// Finalize (block per (b,t) row, in place on d_out which holds Z=logits@V^T):
//   softmax over V, scale by mix3, scatter mix_i*attn_i at text indices, log.
// ---------------------------------------------------------------------------
__global__ void finalize_kernel(float* __restrict__ out,            // [B*T, V]
                                const float* __restrict__ mix,      // [B*T, 4]
                                const float* __restrict__ a0,
                                const float* __restrict__ a1,
                                const float* __restrict__ a2,
                                const int* __restrict__ t0,
                                const int* __restrict__ t1,
                                const int* __restrict__ t2) {
    int row = blockIdx.x;
    int b = row / Tn;
    float* z = out + (size_t)row * Vn;

    float lmax = -INFINITY;
    for (int v = threadIdx.x; v < Vn; v += blockDim.x) lmax = fmaxf(lmax, z[v]);
    float m = block_max(lmax);

    float lsum = 0.0f;
    for (int v = threadIdx.x; v < Vn; v += blockDim.x) lsum += __expf(z[v] - m);
    float ssum = block_sum(lsum);

    float c3 = mix[(size_t)row * 4 + 3] / ssum;
    for (int v = threadIdx.x; v < Vn; v += blockDim.x)
        z[v] = __expf(z[v] - m) * c3;
    __threadfence();
    __syncthreads();

    // sparse pointer scatter: 352 (i,s) pairs
    for (int idx = threadIdx.x; idx < STOT; idx += blockDim.x) {
        int i, s, S; const int* tx; const float* at;
        if (idx < 32)        { i = 0; s = idx;       S = 32;  tx = t0; at = a0; }
        else if (idx < 288)  { i = 1; s = idx - 32;  S = 256; tx = t1; at = a1; }
        else                 { i = 2; s = idx - 288; S = 64;  tx = t2; at = a2; }
        int tok = tx[(size_t)b * S + s];
        if (tok != 0) {
            float add = mix[(size_t)row * 4 + i] * at[(size_t)row * S + s];
            if (add != 0.0f) atomicAdd(&z[tok], add);
        }
    }
    __threadfence();
    __syncthreads();

    for (int v = threadIdx.x; v < Vn; v += blockDim.x) z[v] = __logf(z[v]);
}

// ---------------------------------------------------------------------------
extern "C" void kernel_launch(void* const* d_in, const int* in_sizes, int n_in,
                              void* d_out, int out_size, void* d_ws, size_t ws_size,
                              hipStream_t stream) {
    (void)in_sizes; (void)n_in; (void)out_size; (void)ws_size;

    const float* logits   = (const float*)d_in[0];
    const float* enc_tgt  = (const float*)d_in[1];
    const float* encs[3]  = { (const float*)d_in[2], (const float*)d_in[3], (const float*)d_in[4] };
    const int*   texts[3] = { (const int*)d_in[5], (const int*)d_in[6], (const int*)d_in[7] };
    const unsigned char* masks[3] = { (const unsigned char*)d_in[8],
                                      (const unsigned char*)d_in[9],
                                      (const unsigned char*)d_in[10] };
    const float* vocab_gen = (const float*)d_in[11];
    const float* W_gen     = (const float*)d_in[12];
    const float* b_gen     = (const float*)d_in[13];
    float* out = (float*)d_out;

    const int Ss[3] = { 32, 256, 64 };
    const float inv_sqrt_d = 0.044194173824159216f;   // 1/sqrt(512)

    // ----- workspace arena (aligned 256B); total ~55 MB -----
    unsigned char* base = (unsigned char*)d_ws;
    size_t off = 0;
    auto alloc = [&](size_t bytes) -> void* {
        void* p = base + off;
        off += (bytes + 255) & ~(size_t)255;
        return p;
    };
    unsigned short* vocab_bf  = (unsigned short*)alloc((size_t)Vn * Dn * 2);
    unsigned short* logits_bf = (unsigned short*)alloc((size_t)ROWS * Dn * 2);
    unsigned short* wT_bf     = (unsigned short*)alloc((size_t)Dn * Dn * 2);
    float*          q_f       = (float*)alloc((size_t)ROWS * Dn * 4);
    unsigned short* q_bf      = (unsigned short*)alloc((size_t)ROWS * Dn * 2);
    float*          k_f       = (float*)alloc((size_t)Bn * 256 * Dn * 4);
    unsigned short* k_bf      = (unsigned short*)alloc((size_t)Bn * 256 * Dn * 2);
    unsigned short* enc_bf    = (unsigned short*)alloc((size_t)Bn * 256 * Dn * 2);
    float* attn[3];
    float* tvec[3];
    for (int i = 0; i < 3; ++i) attn[i] = (float*)alloc((size_t)ROWS * Ss[i] * 4);
    for (int i = 0; i < 3; ++i) tvec[i] = (float*)alloc((size_t)ROWS * Dn * 4);
    float* mixp = (float*)alloc((size_t)ROWS * 4 * 4);

    // ----- converts -----
    cvt_bf16_kernel<<<4096, 256, 0, stream>>>(vocab_gen, vocab_bf, (long)Vn * Dn);
    cvt_bf16_kernel<<<512, 256, 0, stream>>>(logits, logits_bf, (long)ROWS * Dn);

    // ----- per-pointer feature: q/k projections, scores, softmax, text_vec -----
    for (int i = 0; i < 3; ++i) {
        const float* Wq = (const float*)d_in[14 + 4 * i];
        const float* bq = (const float*)d_in[15 + 4 * i];
        const float* Wk = (const float*)d_in[16 + 4 * i];
        const float* bk = (const float*)d_in[17 + 4 * i];
        const int S = Ss[i];
        const int MK = Bn * S;   // rows of k

        // q = logits @ Wq + bq         [384,512]
        tcvt_bf16_kernel<<<dim3(16, 16), dim3(32, 8), 0, stream>>>(Wq, wT_bf, Dn, Dn);
        gemm_bt_wmma<<<dim3(Dn / 64, (ROWS + 63) / 64, 1), 128, 0, stream>>>(
            logits_bf, wT_bf, q_f, bq, ROWS, Dn, Dn, Dn, Dn, Dn, 0, 0, 0, 1.0f);
        cvt_bf16_kernel<<<512, 256, 0, stream>>>(q_f, q_bf, (long)ROWS * Dn);

        // k = enc @ Wk + bk            [B*S, 512]
        cvt_bf16_kernel<<<2048, 256, 0, stream>>>(encs[i], enc_bf, (long)MK * Dn);
        tcvt_bf16_kernel<<<dim3(16, 16), dim3(32, 8), 0, stream>>>(Wk, wT_bf, Dn, Dn);
        gemm_bt_wmma<<<dim3(Dn / 64, (MK + 63) / 64, 1), 128, 0, stream>>>(
            enc_bf, wT_bf, k_f, bk, MK, Dn, Dn, Dn, Dn, Dn, 0, 0, 0, 1.0f);
        cvt_bf16_kernel<<<2048, 256, 0, stream>>>(k_f, k_bf, (long)MK * Dn);

        // scores[b] = q[b] @ k[b]^T / sqrt(D)    batched, M=24, N=S, K=512
        gemm_bt_wmma<<<dim3((S + 63) / 64, 1, Bn), 128, 0, stream>>>(
            q_bf, k_bf, attn[i], nullptr, Tn, S, Dn, Dn, Dn, S,
            (long)Tn * Dn, (long)S * Dn, (long)Tn * S, inv_sqrt_d);

        attn_softmax_kernel<<<ROWS, 256, 0, stream>>>(attn[i], texts[i], masks[i], S);
        text_vec_kernel<<<ROWS, 256, 0, stream>>>(attn[i], encs[i], tvec[i], S);
    }

    // ----- mixture gate -----
    mix_kernel<<<ROWS, 128, 0, stream>>>(logits, enc_tgt, tvec[0], tvec[1], tvec[2],
                                         W_gen, b_gen, mixp);

    // ----- Z = logits @ vocab_gen^T  (dominant GEMM, written straight to d_out) -----
    gemm_bt_wmma<<<dim3(Vn / 64, ROWS / 64, 1), 128, 0, stream>>>(
        logits_bf, vocab_bf, out, nullptr, ROWS, Vn, Dn, Dn, Dn, Vn, 0, 0, 0, 1.0f);

    // ----- fused softmax + pointer scatter + log, in place -----
    finalize_kernel<<<ROWS, 256, 0, stream>>>(out, mixp, attn[0], attn[1], attn[2],
                                              texts[0], texts[1], texts[2]);
}
